// LinearAttentionMixer_51522427683043
// MI455X (gfx1250) — matrix operand order repaired and checked by reading, not compiled, not run on hardware
//
#include <hip/hip_runtime.h>

#define BATCH   4
#define SEQ     4096
#define DMODEL  1024
#define NHEAD   16
#define HD      64
#define CHUNKN  128
#define NCHUNK  (SEQ / CHUNKN)        // 32
#define ROWS    (BATCH * SEQ)         // 16384

typedef __attribute__((ext_vector_type(16))) __bf16 v16bf;
typedef __attribute__((ext_vector_type(8)))  float  v8f;

__device__ __forceinline__ v8f wmma_bf16(v16bf a, v16bf b, v8f c) {
  // D = A(16x32 bf16) * B(32x16 bf16) + C(16x16 f32)
  return __builtin_amdgcn_wmma_f32_16x16x32_bf16(false, a, false, b, (short)0, c,
                                                 false, false);
}

// ---- fragment builders (contiguous k order, same in A and B) -------------------------------
// A (16x32, 16-bit): lane holds row M=lane%16.
//   lanes 0-15 : e0..15 -> K=k0+0..15
//   lanes 16-31: e0..15 -> K=k0+16..31
__device__ __forceinline__ v16bf a_frag_bf(const __bf16* rowp, int k0, int kh) {
  v16bf a;
#pragma unroll
  for (int j = 0; j < 16; ++j) a[j] = rowp[k0 + 16 * kh + j];
  return a;
}
// B (32x16, 16-bit): lane holds column N=lane%16; elements K=k0+16*(lane/16)+j,
// sourced from a contiguous run (colp = row of the transposed operand).
__device__ __forceinline__ v16bf b_frag_bf(const __bf16* colp, int k0, int kh) {
  v16bf b;
#pragma unroll
  for (int j = 0; j < 16; ++j) b[j] = colp[k0 + 16 * kh + j];
  return b;
}
__device__ __forceinline__ v16bf b_frag_f32(const float* colp, int k0, int kh) {
  v16bf b;
#pragma unroll
  for (int j = 0; j < 16; ++j) b[j] = (__bf16)colp[k0 + 16 * kh + j];
  return b;
}

// ---- async global -> LDS copy (CDNA5, tracked by ASYNCcnt) ----------------
// LDS offset == low 32 bits of the flat address of a __shared__ object
// (aperture occupies addr[63:32]; ISA 10.2).
__device__ __forceinline__ void async_ld_b128(void* lds_dst, const void* gsrc) {
  unsigned off = (unsigned)(size_t)lds_dst;
  asm volatile("global_load_async_to_lds_b128 %0, %1, off"
               :: "v"(off), "v"(gsrc) : "memory");
}
__device__ __forceinline__ void wait_async0() {
  asm volatile("s_wait_asynccnt 0x0" ::: "memory");
}

// ---- kernel 0: f32 -> bf16 conversion pass --------------------------------
__global__ __launch_bounds__(256) void cvt_f32_bf16_kernel(
    const float* __restrict__ src, unsigned short* __restrict__ dstw, int n4) {
  __bf16* dst = (__bf16*)dstw;
  const int stride = gridDim.x * 256;
  for (int i = blockIdx.x * 256 + threadIdx.x; i < n4; i += stride) {
    const float4 v = ((const float4*)src)[i];
    dst[4 * i + 0] = (__bf16)v.x;
    dst[4 * i + 1] = (__bf16)v.y;
    dst[4 * i + 2] = (__bf16)v.z;
    dst[4 * i + 3] = (__bf16)v.w;
  }
}

// ---- kernel 1: P = feature(x @ W^T), bf16 in -> bf16 out in (b,h,l,d) -----
// block = 8 waves; wave tile = 32(M) x 64(N); block tile = 256 x 64
__global__ __launch_bounds__(256) void proj_qkv_kernel(
    const unsigned short* __restrict__ xw, const unsigned short* __restrict__ Ww,
    unsigned short* __restrict__ outw, int feature) {
  const __bf16* x = (const __bf16*)xw;
  const __bf16* W = (const __bf16*)Ww;
  __bf16* out = (__bf16*)outw;
  const int w = threadIdx.x >> 5;
  const int lane = threadIdx.x & 31;
  const int n = lane & 15, kh = lane >> 4;
  const int mbase = blockIdx.y * 256 + 32 * w;
  const int nbase = blockIdx.x * 64;
  const __bf16* ap0 = x + (size_t)(mbase + n) * DMODEL;
  const __bf16* ap1 = x + (size_t)(mbase + 16 + n) * DMODEL;
  const __bf16* bp0 = W + (size_t)(nbase + 0  + n) * DMODEL;  // B col n == W row n
  const __bf16* bp1 = W + (size_t)(nbase + 16 + n) * DMODEL;
  const __bf16* bp2 = W + (size_t)(nbase + 32 + n) * DMODEL;
  const __bf16* bp3 = W + (size_t)(nbase + 48 + n) * DMODEL;
  v8f acc[2][4] = {};
  for (int k0 = 0; k0 < DMODEL; k0 += 32) {
    __builtin_prefetch(ap0 + k0 + 256, 0, 1);
    __builtin_prefetch(ap1 + k0 + 256, 0, 1);
    v16bf a0 = a_frag_bf(ap0, k0, kh);
    v16bf a1 = a_frag_bf(ap1, k0, kh);
    v16bf b;
    b = b_frag_bf(bp0, k0, kh);
    acc[0][0] = wmma_bf16(a0, b, acc[0][0]);
    acc[1][0] = wmma_bf16(a1, b, acc[1][0]);
    b = b_frag_bf(bp1, k0, kh);
    acc[0][1] = wmma_bf16(a0, b, acc[0][1]);
    acc[1][1] = wmma_bf16(a1, b, acc[1][1]);
    b = b_frag_bf(bp2, k0, kh);
    acc[0][2] = wmma_bf16(a0, b, acc[0][2]);
    acc[1][2] = wmma_bf16(a1, b, acc[1][2]);
    b = b_frag_bf(bp3, k0, kh);
    acc[0][3] = wmma_bf16(a0, b, acc[0][3]);
    acc[1][3] = wmma_bf16(a1, b, acc[1][3]);
  }
#pragma unroll
  for (int af = 0; af < 2; ++af) {
#pragma unroll
    for (int t = 0; t < 4; ++t) {
      const int c = nbase + 16 * t + n;
      const int h = c >> 6, dd = c & 63;
#pragma unroll
      for (int v = 0; v < 8; ++v) {
        const int r = mbase + 16 * af + v + 8 * kh;   // D row = v + 8*(lane/16)
        float val = acc[af][t][v];
        if (feature) val = (val > 0.f) ? (val + 1.f) : __expf(val);  // elu(x)+1
        const int bb = r >> 12, l = r & (SEQ - 1);
        out[(((size_t)bb * NHEAD + h) * SEQ + l) * HD + dd] = (__bf16)val;
      }
    }
  }
}

// ---- kernel 2: chunked causal linear attention, one block per (b,h) -------
__global__ __launch_bounds__(256) void linattn_kernel(
    const unsigned short* __restrict__ Qw, const unsigned short* __restrict__ Kw,
    const unsigned short* __restrict__ Vw, unsigned short* __restrict__ Aw) {
  const __bf16* Q = (const __bf16*)Qw;
  const __bf16* K = (const __bf16*)Kw;
  const __bf16* V = (const __bf16*)Vw;
  __bf16* attn = (__bf16*)Aw;

  __shared__ __bf16 qb[CHUNKN * HD];      // 16 KB  q chunk (C x d)
  __shared__ __bf16 kb[CHUNKN * HD];      // 16 KB  k chunk (C x d)
  __shared__ __bf16 kbT[HD * CHUNKN];     // 16 KB  k^T (d x C)
  __shared__ __bf16 vbT[HD * CHUNKN];     // 16 KB  v^T (d x C)
  __shared__ __bf16 sc[CHUNKN * CHUNKN];  // 32 KB  masked scores (C x C)
  __shared__ float  ST[HD * HD];          // 16 KB  S^T state: ST[e][dd]=S[dd][e]
  __shared__ float  zv[HD];
  __shared__ float  den[CHUNKN];

  const int tid = threadIdx.x;
  const int w = tid >> 5, lane = tid & 31;
  const int n = lane & 15, kh = lane >> 4;
  const int bh = blockIdx.x;               // b*NHEAD + h
  const int bb = bh >> 4, h = bh & 15;
  const __bf16* Qp = Q + (size_t)bh * SEQ * HD;
  const __bf16* Kp = K + (size_t)bh * SEQ * HD;
  const __bf16* Vp = V + (size_t)bh * SEQ * HD;

  for (int i = tid; i < HD * HD; i += 256) ST[i] = 0.f;
  if (tid < HD) zv[tid] = 0.f;
  __syncthreads();

  for (int t = 0; t < NCHUNK; ++t) {
    // -- stage chunk into LDS: q via async DMA, k/v via regs (+ transposes) --
    const __bf16* q0 = Qp + (size_t)t * CHUNKN * HD;
    const __bf16* k0p = Kp + (size_t)t * CHUNKN * HD;
    const __bf16* v0p = Vp + (size_t)t * CHUNKN * HD;
#pragma unroll
    for (int r = 0; r < 4; ++r) {          // 4 x (256 thr x 16B) = 16 KB
      const int i = (tid + r * 256) * 8;   // 8 bf16 per b128
      async_ld_b128(&qb[i], q0 + i);
    }
    for (int i = tid; i < CHUNKN * HD; i += 256) {
      const int c = i >> 6, dd = i & 63;
      __bf16 kv = k0p[i]; kb[i] = kv; kbT[dd * CHUNKN + c] = kv;
      vbT[dd * CHUNKN + c] = v0p[i];
    }
    wait_async0();
    __syncthreads();

    // -- scores = tril(qb @ kb^T): wave w owns C-rows [16w,16w+16) --
    const __bf16* qrow = qb + (16 * w + n) * HD;
    for (int sj = w + 1; sj < 8; ++sj) {      // fully-masked tiles -> 0
#pragma unroll
      for (int v = 0; v < 8; ++v)
        sc[(16 * w + v + 8 * kh) * CHUNKN + 16 * sj + n] = (__bf16)0.f;
    }
    for (int sj = 0; sj <= w; ++sj) {
      const __bf16* krow = kb + (16 * sj + n) * HD;   // B col s = kb row s
      v8f acc = {};
      acc = wmma_bf16(a_frag_bf(qrow, 0, kh),  b_frag_bf(krow, 0, kh),  acc);
      acc = wmma_bf16(a_frag_bf(qrow, 32, kh), b_frag_bf(krow, 32, kh), acc);
      if (sj == w) {                          // only diagonal tile needs the mask
#pragma unroll
        for (int v = 0; v < 8; ++v) {
          const int c = 16 * w + v + 8 * kh, s = 16 * sj + n;
          sc[c * CHUNKN + s] = (__bf16)((c >= s) ? acc[v] : 0.f);
        }
      } else {
#pragma unroll
        for (int v = 0; v < 8; ++v)
          sc[(16 * w + v + 8 * kh) * CHUNKN + 16 * sj + n] = (__bf16)acc[v];
      }
    }
    __syncthreads();

    // -- den[c] = qb[c]·z + sum_s sc[c][s] --
    if (tid < CHUNKN) {
      float s1 = 0.f;
      for (int dd = 0; dd < HD; ++dd) s1 += (float)qb[tid * HD + dd] * zv[dd];
      float s2 = 0.f;
      for (int s = 0; s < CHUNKN; ++s) s2 += (float)sc[tid * CHUNKN + s];
      den[tid] = s1 + s2;
    }
    __syncthreads();

    // -- num = qb @ S + sc @ vb ; out = num/den + eps --
    const __bf16* srow = sc + (16 * w + n) * CHUNKN;
    for (int ej = 0; ej < 4; ++ej) {
      v8f acc = {};
      const float* STrow = ST + (16 * ej + n) * HD;       // B[dd][e]=ST[e][dd]
      acc = wmma_bf16(a_frag_bf(qrow, 0, kh),  b_frag_f32(STrow, 0, kh),  acc);
      acc = wmma_bf16(a_frag_bf(qrow, 32, kh), b_frag_f32(STrow, 32, kh), acc);
      const __bf16* vrow = vbT + (16 * ej + n) * CHUNKN;  // B[s][e]=vbT[e][s]
#pragma unroll
      for (int k0 = 0; k0 < CHUNKN; k0 += 32)
        acc = wmma_bf16(a_frag_bf(srow, k0, kh), b_frag_bf(vrow, k0, kh), acc);
#pragma unroll
      for (int v = 0; v < 8; ++v) {
        const int c = 16 * w + v + 8 * kh;
        const int e = 16 * ej + n;
        const float val = acc[v] / den[c] + 1e-12f;
        const size_t gr = (size_t)bb * SEQ + (size_t)t * CHUNKN + c;
        attn[gr * DMODEL + h * HD + e] = (__bf16)val;
      }
    }
    __syncthreads();

    // -- state update: ST[e][dd] += sum_s vbT[e][s]*kbT[dd][s] (2 tiles/wave) --
    for (int rep = 0; rep < 2; ++rep) {
      const int idx = 2 * w + rep;
      const int ei = idx >> 2, dj = idx & 3;
      v8f acc;
#pragma unroll
      for (int v = 0; v < 8; ++v)
        acc[v] = ST[(16 * ei + v + 8 * kh) * HD + 16 * dj + n];
      const __bf16* arow = vbT + (16 * ei + n) * CHUNKN;
      const __bf16* brow = kbT + (16 * dj + n) * CHUNKN;  // B col dd = kbT row dd
#pragma unroll
      for (int k0 = 0; k0 < CHUNKN; k0 += 32)
        acc = wmma_bf16(a_frag_bf(arow, k0, kh), b_frag_bf(brow, k0, kh), acc);
#pragma unroll
      for (int v = 0; v < 8; ++v)
        ST[(16 * ei + v + 8 * kh) * HD + 16 * dj + n] = acc[v];
    }
    if (tid < HD) {           // z[dd] += sum_c kb[c][dd]
      float s = 0.f;
      for (int c = 0; c < CHUNKN; ++c) s += (float)kbT[tid * CHUNKN + c];
      zv[tid] += s;
    }
    __syncthreads();
  }
}

// ---- kernel 3: out = attn @ Wo^T, bf16 in, f32 out ------------------------
__global__ __launch_bounds__(256) void out_proj_kernel(
    const unsigned short* __restrict__ attnw, const unsigned short* __restrict__ Wow,
    float* __restrict__ out) {
  const __bf16* attn = (const __bf16*)attnw;
  const __bf16* Wo = (const __bf16*)Wow;
  const int w = threadIdx.x >> 5;
  const int lane = threadIdx.x & 31;
  const int n = lane & 15, kh = lane >> 4;
  const int mbase = blockIdx.y * 256 + 32 * w;
  const int nbase = blockIdx.x * 64;
  const __bf16* ap0 = attn + (size_t)(mbase + n) * DMODEL;
  const __bf16* ap1 = attn + (size_t)(mbase + 16 + n) * DMODEL;
  const __bf16* bp0 = Wo + (size_t)(nbase + 0  + n) * DMODEL;
  const __bf16* bp1 = Wo + (size_t)(nbase + 16 + n) * DMODEL;
  const __bf16* bp2 = Wo + (size_t)(nbase + 32 + n) * DMODEL;
  const __bf16* bp3 = Wo + (size_t)(nbase + 48 + n) * DMODEL;
  v8f acc[2][4] = {};
  for (int k0 = 0; k0 < DMODEL; k0 += 32) {
    v16bf a0 = a_frag_bf(ap0, k0, kh);
    v16bf a1 = a_frag_bf(ap1, k0, kh);
    v16bf b;
    b = b_frag_bf(bp0, k0, kh);
    acc[0][0] = wmma_bf16(a0, b, acc[0][0]);
    acc[1][0] = wmma_bf16(a1, b, acc[1][0]);
    b = b_frag_bf(bp1, k0, kh);
    acc[0][1] = wmma_bf16(a0, b, acc[0][1]);
    acc[1][1] = wmma_bf16(a1, b, acc[1][1]);
    b = b_frag_bf(bp2, k0, kh);
    acc[0][2] = wmma_bf16(a0, b, acc[0][2]);
    acc[1][2] = wmma_bf16(a1, b, acc[1][2]);
    b = b_frag_bf(bp3, k0, kh);
    acc[0][3] = wmma_bf16(a0, b, acc[0][3]);
    acc[1][3] = wmma_bf16(a1, b, acc[1][3]);
  }
#pragma unroll
  for (int af = 0; af < 2; ++af) {
#pragma unroll
    for (int t = 0; t < 4; ++t) {
#pragma unroll
      for (int v = 0; v < 8; ++v) {
        const int r = mbase + 16 * af + v + 8 * kh;
        const int c = nbase + 16 * t + n;
        out[(size_t)r * DMODEL + c] = acc[af][t][v];
      }
    }
  }
}

extern "C" void kernel_launch(void* const* d_in, const int* in_sizes, int n_in,
                              void* d_out, int out_size, void* d_ws, size_t ws_size,
                              hipStream_t stream) {
  (void)in_sizes; (void)n_in; (void)out_size; (void)ws_size;
  const float* x  = (const float*)d_in[0];
  const float* Wq = (const float*)d_in[1];
  const float* Wk = (const float*)d_in[2];
  const float* Wv = (const float*)d_in[3];
  const float* Wo = (const float*)d_in[4];

  // workspace layout (bf16):
  //   Xb 32MB | Wq/Wk/Wv/Wo 2MB each | Qb/Kb/Vb 32MB each | Ab 32MB  = 168MB
  const size_t XSZ = (size_t)ROWS * DMODEL * 2;     // 32 MB
  const size_t WSZ = (size_t)DMODEL * DMODEL * 2;   // 2 MB
  char* ws = (char*)d_ws;
  unsigned short* Xb  = (unsigned short*)(ws);
  unsigned short* Wqb = (unsigned short*)(ws + XSZ + 0 * WSZ);
  unsigned short* Wkb = (unsigned short*)(ws + XSZ + 1 * WSZ);
  unsigned short* Wvb = (unsigned short*)(ws + XSZ + 2 * WSZ);
  unsigned short* Wob = (unsigned short*)(ws + XSZ + 3 * WSZ);
  unsigned short* Qb  = (unsigned short*)(ws + XSZ + 4 * WSZ + 0 * XSZ);
  unsigned short* Kb  = (unsigned short*)(ws + XSZ + 4 * WSZ + 1 * XSZ);
  unsigned short* Vb  = (unsigned short*)(ws + XSZ + 4 * WSZ + 2 * XSZ);
  unsigned short* Ab  = (unsigned short*)(ws + XSZ + 4 * WSZ + 3 * XSZ);

  // conversion pass (everything is L2-resident afterwards: 192MB L2)
  cvt_f32_bf16_kernel<<<4096, 256, 0, stream>>>(x,  Xb,  ROWS * DMODEL / 4);
  cvt_f32_bf16_kernel<<<1024, 256, 0, stream>>>(Wq, Wqb, DMODEL * DMODEL / 4);
  cvt_f32_bf16_kernel<<<1024, 256, 0, stream>>>(Wk, Wkb, DMODEL * DMODEL / 4);
  cvt_f32_bf16_kernel<<<1024, 256, 0, stream>>>(Wv, Wvb, DMODEL * DMODEL / 4);
  cvt_f32_bf16_kernel<<<1024, 256, 0, stream>>>(Wo, Wob, DMODEL * DMODEL / 4);

  dim3 gproj(DMODEL / 64, ROWS / 256);   // (16, 64)
  proj_qkv_kernel<<<gproj, 256, 0, stream>>>(Xb, Wqb, Qb, 1);
  proj_qkv_kernel<<<gproj, 256, 0, stream>>>(Xb, Wkb, Kb, 1);
  proj_qkv_kernel<<<gproj, 256, 0, stream>>>(Xb, Wvb, Vb, 0);
  linattn_kernel<<<BATCH * NHEAD, 256, 0, stream>>>(Qb, Kb, Vb, Ab);
  out_proj_kernel<<<gproj, 256, 0, stream>>>(Ab, Wob, (float*)d_out);
}
